// GraphConvolution_76355928588647
// MI455X (gfx1250) — compile-verified, matrix operand order checked by reading, not run on hardware
//
#include <hip/hip_runtime.h>
#include <hip/hip_bf16.h>

typedef __attribute__((ext_vector_type(16))) __bf16 v16bf;
typedef __attribute__((ext_vector_type(8)))  float  v8f;

#define BB 4
#define NN 4096
#define DD 256
#define FF 256

__device__ __forceinline__ unsigned short f2bf(float f) {
    unsigned int u = __builtin_bit_cast(unsigned int, f);
    unsigned int r = u + 0x7fffu + ((u >> 16) & 1u);
    return (unsigned short)(r >> 16);
}

__device__ __forceinline__ v16bf ld_frag(const unsigned short* p0, const unsigned short* p1) {
    union { v16bf v; uint4 u[2]; } r;
    r.u[0] = *(const uint4*)p0;
    r.u[1] = *(const uint4*)p1;
    return r.v;
}

// ---------------- K1: x (fp32) -> xb (bf16) ----------------
__global__ void cvt_x_kernel(const float* __restrict__ x, unsigned short* __restrict__ xb) {
    size_t i = (size_t)blockIdx.x * 256 + threadIdx.x;
    xb[i] = f2bf(x[i]);
}

// ---------------- K1b: weight -> weightT (bf16) ----------------
__global__ void cvt_w_kernel(const float* __restrict__ w, unsigned short* __restrict__ wT) {
    int i = blockIdx.x * 256 + threadIdx.x;   // i = f*256 + d
    int d = i & 255, f = i >> 8;
    wT[i] = f2bf(w[d * 256 + f]);
}

// ---------------- K2: supportT[b][f][m] = (x @ W)^T in bf16 ----------------
// 4 waves/block, each wave computes one 16x16 tile over K=256 (8 WMMAs).
__global__ __launch_bounds__(128) void support_kernel(const unsigned short* __restrict__ xb,
                                                      const unsigned short* __restrict__ wT,
                                                      unsigned short* __restrict__ st) {
    int lane = threadIdx.x & 31;
    int wid  = threadIdx.x >> 5;
    int gw   = blockIdx.x * 4 + wid;          // 16384 tiles total
    int tile_m = gw >> 4;                     // 0..1023 (rows of B*N)
    int tile_f = gw & 15;                     // 0..15
    int lo = lane & 15, hi = lane >> 4;

    const unsigned short* arow = xb + ((size_t)(tile_m * 16 + lo)) * DD;
    const unsigned short* brow = wT + ((size_t)(tile_f * 16 + lo)) * DD;

    v8f acc = {};
#pragma unroll
    for (int kk = 0; kk < 8; ++kk) {
        v16bf a = ld_frag(arow + kk * 32 + hi * 8, arow + kk * 32 + hi * 8 + 16);
        v16bf b = ld_frag(brow + kk * 32 + hi * 8, brow + kk * 32 + hi * 8 + 16);
        acc = __builtin_amdgcn_wmma_f32_16x16x32_bf16(false, a, false, b, (short)0, acc, false, false);
    }
    // store transposed: supportT[b][f][m], 8 consecutive m per lane -> one b128
    int g0 = tile_m * 16 + 8 * hi;            // global row (over B*N)
    int bidx = g0 >> 12;
    int mloc = g0 & (NN - 1);
    int f = tile_f * 16 + lo;
    unsigned short tmp[8];
#pragma unroll
    for (int r = 0; r < 8; ++r) tmp[r] = f2bf(acc[r]);
    *(uint4*)(st + ((size_t)(bidx * FF + f)) * NN + mloc) = *(const uint4*)tmp;
}

// ---------------- K3: fused  softmax(xx^T)*mask @ support  + bias, ReLU, transpose ----------------
// 4 waves per block; each wave owns every 4th 32-key tile (split-K over the softmax axis),
// keeping a private 16x256 fp32 accumulator + unmasked denominator; partials merged via LDS.
__global__ __launch_bounds__(128) void attn_kernel(const float* __restrict__ mask,
                                                   const float* __restrict__ bias,
                                                   const unsigned short* __restrict__ xb,
                                                   const unsigned short* __restrict__ st,
                                                   float* __restrict__ out) {
    const int lane = threadIdx.x & 31;
    const int wid  = threadIdx.x >> 5;
    const int lo = lane & 15, hi = lane >> 4;
    const int rt = blockIdx.x & 255;          // N/16 row tiles
    const int b  = blockIdx.x >> 8;
    const int rowbase = rt * 16;

    __shared__ __align__(16) unsigned short pshare[4][16 * 32]; // per-wave P staging
    __shared__ float accbuf[16 * 256];                          // 16 KB merge buffer
    __shared__ float denbuf[16];

    v8f acc[16];
#pragma unroll
    for (int i = 0; i < 16; ++i) acc[i] = (v8f){};
    float den[8] = {0.f, 0.f, 0.f, 0.f, 0.f, 0.f, 0.f, 0.f};

    const unsigned short* qrow = xb + ((size_t)(b * NN + rowbase + lo)) * DD;
    const size_t maskrowbase = ((size_t)b * NN + rowbase) * NN;
    unsigned short* myp = pshare[wid];

    for (int mt = wid; mt < NN / 32; mt += 4) {
        const int mbase = mt * 32;
        // speculative prefetch of this wave's next mask tile (streamed once, 256 MB total)
        if (mt + 4 < NN / 32) {
            __builtin_prefetch(mask + maskrowbase + (size_t)lo * NN + (mt + 4) * 32 + hi * 16, 0, 1);
        }
#pragma unroll
        for (int t = 0; t < 2; ++t) {
            const int cb = mbase + t * 16;
            const unsigned short* krow = xb + ((size_t)(b * NN + cb + lo)) * DD;
            v8f s = {};
#pragma unroll
            for (int kk = 0; kk < 8; ++kk) {
                v16bf a  = ld_frag(qrow + kk * 32 + hi * 8, qrow + kk * 32 + hi * 8 + 16);
                v16bf kb = ld_frag(krow + kk * 32 + hi * 8, krow + kk * 32 + hi * 8 + 16);
                s = __builtin_amdgcn_wmma_f32_16x16x32_bf16(false, a, false, kb, (short)0, s, false, false);
            }
            const int col = cb + lo;
#pragma unroll
            for (int r = 0; r < 8; ++r) {
                float p = __expf(s[r]);            // scores ~O(1): no max-shift needed
                den[r] += p;                        // denominator is UNmasked (mask applied after softmax)
                float mv = mask[maskrowbase + (size_t)(r + 8 * hi) * NN + col];
                myp[(r + 8 * hi) * 32 + t * 16 + lo] = f2bf(p * mv);
            }
        }
        // producer == consumer wave: LDS is in-order per wave; wait + compiler barrier for safety
        asm volatile("s_wait_dscnt 0x0" ::: "memory");
        v16bf pf;
        {
            union { v16bf v; uint4 u[2]; } r;
            r.u[0] = *(const uint4*)(&myp[lo * 32 + hi * 8]);
            r.u[1] = *(const uint4*)(&myp[lo * 32 + hi * 8 + 16]);
            pf = r.v;
        }
#pragma unroll
        for (int ft = 0; ft < 16; ++ft) {
            const int f = ft * 16 + lo;
            const unsigned short* vp = st + ((size_t)(b * FF + f)) * NN + mbase + hi * 8;
            v16bf vf = ld_frag(vp, vp + 16);
            acc[ft] = __builtin_amdgcn_wmma_f32_16x16x32_bf16(false, pf, false, vf, (short)0, acc[ft], false, false);
        }
        asm volatile("" ::: "memory");  // keep next iteration's stores after this iteration's reads
    }

    // row-sum of this wave's denominator partial across the 16 lanes sharing a row group
#pragma unroll
    for (int r = 0; r < 8; ++r) {
        float d = den[r];
        d += __shfl_xor(d, 1);
        d += __shfl_xor(d, 2);
        d += __shfl_xor(d, 4);
        d += __shfl_xor(d, 8);
        den[r] = d;
    }

    // merge the 4 split-K partials through LDS
    __syncthreads();
    if (wid == 0) {
#pragma unroll
        for (int ft = 0; ft < 16; ++ft)
#pragma unroll
            for (int r = 0; r < 8; ++r)
                accbuf[(r + 8 * hi) * 256 + ft * 16 + lo] = acc[ft][r];
        if (lo == 0) {
#pragma unroll
            for (int r = 0; r < 8; ++r) denbuf[r + 8 * hi] = den[r];
        }
    }
    __syncthreads();
    if (wid != 0) {
#pragma unroll
        for (int ft = 0; ft < 16; ++ft)
#pragma unroll
            for (int r = 0; r < 8; ++r)
                atomicAdd(&accbuf[(r + 8 * hi) * 256 + ft * 16 + lo], acc[ft][r]);
        if (lo == 0) {
#pragma unroll
            for (int r = 0; r < 8; ++r) atomicAdd(&denbuf[r + 8 * hi], den[r]);
        }
    }
    __syncthreads();

    float rden[8];
#pragma unroll
    for (int r = 0; r < 8; ++r) rden[r] = __builtin_amdgcn_rcpf(denbuf[r + 8 * hi]);

    // out[b][f][n] = relu(acc/den + bias[f]); each wave writes 4 f-tiles
#pragma unroll
    for (int i = 0; i < 4; ++i) {
        const int ft = wid * 4 + i;
        const int f = ft * 16 + lo;
        const float bi = bias[f];
        float* op = out + ((size_t)(b * FF + f)) * NN + rowbase + 8 * hi;
#pragma unroll
        for (int r = 0; r < 8; ++r) {
            float v = accbuf[(r + 8 * hi) * 256 + ft * 16 + lo] * rden[r] + bi;
            op[r] = v > 0.f ? v : 0.f;
        }
    }
}

// ---------------- K4: per-channel mean / inv-std over (B, N) ----------------
__global__ __launch_bounds__(256) void stats_kernel(const float* __restrict__ out, float* __restrict__ stats) {
    const int f = blockIdx.x, t = threadIdx.x;
    float s = 0.f, sq = 0.f;
    for (int b = 0; b < BB; ++b) {
        const float* p = out + ((size_t)(b * FF + f)) * NN;
        for (int i = t; i < NN; i += 256) { float v = p[i]; s += v; sq += v * v; }
    }
    __shared__ float ssum[256], ssq[256];
    ssum[t] = s; ssq[t] = sq;
    __syncthreads();
    for (int off = 128; off > 0; off >>= 1) {
        if (t < off) { ssum[t] += ssum[t + off]; ssq[t] += ssq[t + off]; }
        __syncthreads();
    }
    if (t == 0) {
        const float inv_n = 1.0f / (float)(BB * NN);
        float mean = ssum[0] * inv_n;
        float var  = ssq[0] * inv_n - mean * mean;
        stats[f] = mean;
        stats[FF + f] = rsqrtf(var + 1e-5f);
    }
}

// ---------------- K5: in-place BN affine ----------------
__global__ void bn_kernel(float* __restrict__ out, const float* __restrict__ stats,
                          const float* __restrict__ gamma, const float* __restrict__ beta) {
    size_t i = (size_t)blockIdx.x * 256 + threadIdx.x;
    int f = (int)((i >> 12) & (FF - 1));
    float v = out[i];
    out[i] = (v - stats[f]) * stats[FF + f] * gamma[f] + beta[f];
}

extern "C" void kernel_launch(void* const* d_in, const int* in_sizes, int n_in,
                              void* d_out, int out_size, void* d_ws, size_t ws_size,
                              hipStream_t stream) {
    const float* x      = (const float*)d_in[0];
    const float* mask   = (const float*)d_in[1];
    const float* weight = (const float*)d_in[2];
    const float* bias   = (const float*)d_in[3];
    const float* gamma  = (const float*)d_in[4];
    const float* beta   = (const float*)d_in[5];
    float* out = (float*)d_out;

    char* ws = (char*)d_ws;
    unsigned short* xb = (unsigned short*)(ws);                         // 8 MB  bf16 x
    unsigned short* wT = (unsigned short*)(ws + 8388608);               // 128KB bf16 W^T
    unsigned short* st = (unsigned short*)(ws + 8388608 + 131072);      // 8 MB  bf16 support^T
    float* stats       = (float*)(ws + 8388608 + 131072 + 8388608);     // 2KB

    cvt_x_kernel<<<(BB * NN * DD) / 256, 256, 0, stream>>>(x, xb);
    cvt_w_kernel<<<(DD * FF) / 256, 256, 0, stream>>>(weight, wT);
    support_kernel<<<(BB * NN / 16) * (FF / 16) / 4, 128, 0, stream>>>(xb, wT, st);
    attn_kernel<<<BB * (NN / 16), 128, 0, stream>>>(mask, bias, xb, st, out);
    stats_kernel<<<FF, 256, 0, stream>>>(out, stats);
    bn_kernel<<<(BB * FF * NN) / 256, 256, 0, stream>>>(out, stats, gamma, beta);
}
